// Generator_18915035972391
// MI455X (gfx1250) — matrix operand order repaired and checked by reading, not compiled, not run on hardware
//
#include <hip/hip_runtime.h>
#include <hip/hip_bf16.h>

typedef __attribute__((ext_vector_type(16))) _Float16 v16h;
typedef __attribute__((ext_vector_type(8)))  _Float16 v8h;
typedef __attribute__((ext_vector_type(8)))  float    v8f;

#define NB 16  // batch

// ---------------------------------------------------------------------------
// Fused conv3x3(pad=1) + BN(scale/shift) + optional residual add + optional
// ReLU, implemented as implicit GEMM on v_wmma_f32_16x16x32_f16.
// One wave = 16 out-channels x 16 pixels tile. Block = 4 waves = 64 pixels of
// one output row. K = Cin*9 (zero-padded to multiple of 32).
// Staging pass builds the im2col B panel [64 px][Kpad] in LDS (pixel-major,
// padded row stride) so the hot loop is pure 128-bit loads + WMMA.
// Weights are pre-packed [Opad][Kpad] fp16, zero padded, so A loads need no
// bounds checks (no exec-mask divergence in the K loop).
// ---------------------------------------------------------------------------
__global__ void __launch_bounds__(128)
conv3x3_wmma(const float* __restrict__ in, int C, int H, int W,
             const _Float16* __restrict__ wp, int O, int Kpad,
             const float* __restrict__ scale, const float* __restrict__ shift,
             const float* __restrict__ res, int resCt, int resB,
             float* __restrict__ out, int outCt, int outB, int relu)
{
    extern __shared__ _Float16 smem[];       // [64][KROW] im2col panel
    const int KROW = Kpad + 8;               // +16B: bank spread, keeps 16B align
    const int n   = blockIdx.z;
    const int h   = blockIdx.y;
    const int w0  = blockIdx.x * 64;
    const int tid = threadIdx.x;

    // ---- Stage im2col panel: k = c*9 + r*3 + s, px = 0..63.
    // Consecutive tid -> consecutive px -> coalesced global reads.
    const int CK = C * 9;
    for (int idx = tid; idx < Kpad * 64; idx += 128) {
        int k  = idx >> 6;
        int px = idx & 63;
        float v = 0.f;
        if (k < CK) {
            int c = k / 9, rs = k - c * 9, r = rs / 3, s = rs - r * 3;
            int gr = h - 1 + r, gc = w0 + px + s - 1;
            if ((unsigned)gr < (unsigned)H && (unsigned)gc < (unsigned)W)
                v = in[(((size_t)n * C + c) * H + gr) * W + gc];
        }
        smem[px * KROW + k] = (_Float16)v;
    }
    __syncthreads();

    const int wave = tid >> 5;
    const int lane = tid & 31;
    const bool hi  = lane >= 16;
    const int ln   = lane & 15;
    const int pxl  = wave * 16 + ln;         // local pixel / B column
    const _Float16* brow = smem + (size_t)pxl * KROW;

    for (int o0 = 0; o0 < O; o0 += 16) {
        v8f acc = {};
        // A row for this lane (weights padded to Opad rows -> always valid).
        const _Float16* wrow = wp + (size_t)(o0 + ln) * Kpad;
        for (int k0 = 0; k0 < Kpad; k0 += 32) {
            // A 16x32 f16 layout: lo lanes K k0+{0..7,16..23}, hi lanes +8.
            int ka = k0 + (hi ? 8 : 0);
            v8h a0 = *(const v8h*)(wrow + ka);
            v8h a1 = *(const v8h*)(wrow + ka + 16);
            v16h a = __builtin_shufflevector(a0, a1, 0, 1, 2, 3, 4, 5, 6, 7,
                                             8, 9, 10, 11, 12, 13, 14, 15);
            // B 32x16 f16 layout: lo lanes K k0+{0..7,16..23}, hi lanes +8.
            int kb = k0 + (hi ? 8 : 0);
            v8h b0 = *(const v8h*)(brow + kb);
            v8h b1 = *(const v8h*)(brow + kb + 16);
            v16h b = __builtin_shufflevector(b0, b1, 0, 1, 2, 3, 4, 5, 6, 7,
                                             8, 9, 10, 11, 12, 13, 14, 15);
            acc = __builtin_amdgcn_wmma_f32_16x16x32_f16(
                false, a, false, b, (short)0, acc, false, false);
        }
        // ---- Epilogue: BN + residual + ReLU, write f32.
        // C/D layout: lanes 0-15 hold M=0..7 (v0-7), lanes 16-31 hold M=8..15.
        int px = w0 + pxl;
        #pragma unroll
        for (int r = 0; r < 8; ++r) {
            int m   = (hi ? 8 : 0) + r;
            int och = o0 + m;
            if (och < O) {
                float v = acc[r] * scale[och] + shift[och];
                if (res)
                    v += res[(((size_t)n * resCt + resB + och) * H + h) * W + px];
                if (relu) v = fmaxf(v, 0.f);
                out[(((size_t)n * outCt + outB + och) * H + h) * W + px] = v;
            }
        }
    }
}

// Pack conv weights (OIHW f32, or IOHW flipped for conv-transpose) into a
// row-major fp16 matrix [Opad][Kpad], K = c*9 + r*3 + s, zero padded in both
// dimensions so the GEMM needs no bounds checks.
__global__ void pack_w(const float* __restrict__ w, int O, int Opad, int I,
                       int tr, _Float16* __restrict__ dst, int Kpad)
{
    int idx = blockIdx.x * 256 + threadIdx.x;
    if (idx >= Opad * Kpad) return;
    int o = idx / Kpad;
    int k = idx - o * Kpad;
    float v = 0.f;
    if (o < O && k < I * 9) {
        int c  = k / 9;
        int rs = k - c * 9;
        int r  = rs / 3;
        int s  = rs - r * 3;
        if (tr) v = w[((size_t)c * O + o) * 9 + (2 - r) * 3 + (2 - s)];
        else    v = w[((size_t)o * I + c) * 9 + r * 3 + s];
    }
    dst[idx] = (_Float16)v;
}

// Fold BN (+ optional conv bias) into per-channel scale/shift.
__global__ void bnfold(const float* __restrict__ g, const float* __restrict__ b,
                       const float* __restrict__ m, const float* __restrict__ v,
                       const float* __restrict__ bias,
                       float* __restrict__ sc, float* __restrict__ sh, int Cc)
{
    int i = blockIdx.x * 64 + threadIdx.x;
    if (i >= Cc) return;
    float s = g[i] * rsqrtf(v[i] + 1e-5f);
    sc[i] = s;
    sh[i] = b[i] - m[i] * s + (bias ? bias[i] * s : 0.f);
}

__global__ void avgpool2k(const float* __restrict__ in, float* __restrict__ out,
                          int C, int H, int W)
{
    int Ho = H >> 1, Wo = W >> 1;
    size_t total = (size_t)NB * C * Ho * Wo;
    for (size_t idx = blockIdx.x * (size_t)blockDim.x + threadIdx.x; idx < total;
         idx += (size_t)gridDim.x * blockDim.x) {
        size_t w = idx % Wo; size_t t = idx / Wo;
        size_t h = t % Ho; t /= Ho;
        size_t c = t % C; size_t n = t / C;
        const float* p = in + ((n * C + c) * H + 2 * h) * W + 2 * w;
        out[idx] = 0.25f * (p[0] + p[1] + p[W] + p[W + 1]);
    }
}

// Bilinear x2, align_corners=true.
__global__ void upsample2x(const float* __restrict__ in, float* __restrict__ out,
                           int C, int H, int W)
{
    int Ho = 2 * H, Wo = 2 * W;
    float shh = (float)(H - 1) / (float)(Ho - 1);
    float sww = (float)(W - 1) / (float)(Wo - 1);
    size_t total = (size_t)NB * C * Ho * Wo;
    for (size_t idx = blockIdx.x * (size_t)blockDim.x + threadIdx.x; idx < total;
         idx += (size_t)gridDim.x * blockDim.x) {
        int w = (int)(idx % Wo); size_t t = idx / Wo;
        int h = (int)(t % Ho); t /= Ho;
        int c = (int)(t % C); int n = (int)(t / C);
        float ph = h * shh; int h0 = (int)ph; int h1 = min(h0 + 1, H - 1); float fh = ph - h0;
        float pw = w * sww; int w0 = (int)pw; int w1 = min(w0 + 1, W - 1); float fw = pw - w0;
        const float* base = in + (((size_t)n * C + c) * H) * W;
        float v00 = base[(size_t)h0 * W + w0], v01 = base[(size_t)h0 * W + w1];
        float v10 = base[(size_t)h1 * W + w0], v11 = base[(size_t)h1 * W + w1];
        out[idx] = v00 * (1 - fh) * (1 - fw) + v01 * (1 - fh) * fw
                 + v10 * fh * (1 - fw)       + v11 * fh * fw;
    }
}

struct PW { const _Float16* w; int Kpad; };
struct SS { const float* sc; const float* sh; };

extern "C" void kernel_launch(void* const* d_in, const int* in_sizes, int n_in,
                              void* d_out, int out_size, void* d_ws, size_t ws_size,
                              hipStream_t stream)
{
    (void)in_sizes; (void)n_in; (void)out_size; (void)ws_size;
    const float* X = (const float*)d_in[0];
    const float* Y = (const float*)d_in[1];
    float* OUT = (float*)d_out;

    // ---- workspace carve -------------------------------------------------
    char* ws = (char*)d_ws;
    size_t off = 0;
    auto allocF = [&](size_t n) { float* p = (float*)(ws + off); off += n * 4; return p; };
    float* convfull1 = allocF(16777216);  // [16,16,256,256]
    float* buf16_128 = allocF(4194304);   // [16,16,128,128]
    float* convfull2 = allocF(8388608);   // [16,32,128,128]
    float* pooledB   = allocF(2097152);   // [16,32,64,64]
    float* c3out     = allocF(1048576);   // [16,16,64,64]
    float* t0        = allocF(1048576);
    float* t1        = allocF(1048576);
    float* catb      = allocF(2097152);   // [16,32,64,64]
    float* bigU      = allocF(33554432);  // [16,32,256,256]
    _Float16* wreg = (_Float16*)(ws + off); off += (1u << 21);
    float*    freg = (float*)(ws + off);    off += (1u << 16);
    size_t hoff = 0, foff = 0;

    // ---- helpers ---------------------------------------------------------
    auto packW = [&](int idx, int O, int I, int tr) {
        PW p; p.Kpad = ((I * 9 + 31) / 32) * 32;
        int Opad = ((O + 15) / 16) * 16;
        _Float16* dst = wreg + hoff; hoff += (size_t)Opad * p.Kpad;
        int total = Opad * p.Kpad;
        pack_w<<<(total + 255) / 256, 256, 0, stream>>>(
            (const float*)d_in[idx], O, Opad, I, tr, dst, p.Kpad);
        p.w = dst; return p;
    };
    auto foldBN = [&](int gIdx, int biasIdx, int Cc) {
        float* sc = freg + foff; foff += Cc;
        float* sh = freg + foff; foff += Cc;
        bnfold<<<(Cc + 63) / 64, 64, 0, stream>>>(
            (const float*)d_in[gIdx], (const float*)d_in[gIdx + 1],
            (const float*)d_in[gIdx + 2], (const float*)d_in[gIdx + 3],
            biasIdx >= 0 ? (const float*)d_in[biasIdx] : nullptr, sc, sh, Cc);
        SS s; s.sc = sc; s.sh = sh; return s;
    };
    auto conv = [&](const float* in, int C, int H, int W, PW pw, int O, SS ss,
                    const float* res, int resCt, int resB,
                    float* out, int outCt, int outB, int relu) {
        dim3 g(W / 64, H, NB);
        size_t lds = (size_t)64 * (pw.Kpad + 8) * sizeof(_Float16);
        conv3x3_wmma<<<g, 128, lds, stream>>>(in, C, H, W, pw.w, O, pw.Kpad,
                                              ss.sc, ss.sh, res, resCt, resB,
                                              out, outCt, outB, relu);
    };
    auto pool = [&](const float* in, int C, int H, int W, float* out) {
        size_t total = (size_t)NB * C * (H / 2) * (W / 2);
        size_t b = (total + 255) / 256; if (b > 4096) b = 4096;
        avgpool2k<<<(int)b, 256, 0, stream>>>(in, out, C, H, W);
    };
    auto ups = [&](const float* in, int C, int H, int W, float* out) {
        size_t total = (size_t)NB * C * 4 * H * W;
        size_t b = (total + 255) / 256; if (b > 8192) b = 8192;
        upsample2x<<<(int)b, 256, 0, stream>>>(in, out, C, H, W);
    };

    // Input index map (dict flatten): x=0, y=1; params start at 2.
    // convbn = {w, b, g, b, m, v} (6). blk = {w1, bn1(4), w2, bn2(4)} (10),
    // +{dw, dbn(4)} if down (15).
    // c1x=2 c2x=8 c3x=14 | c1y=20 c2y=26 c3y=32 | b1x=38 b2x=48 b1y=58 b2y=68
    // b1c=78 (dw=88) b2c=93 | up1_w=103 up1_b=104 up1_bn=105 up2_w=109
    // up2_b=110 up2_bn=111

    auto branch = [&](const float* inp, int cbase, int bbase, int catOfs) {
        PW pw; SS ss;
        // conv-bn-relu + pool x2, conv-bn-relu
        pw = packW(cbase + 0, 16, 3, 0);  ss = foldBN(cbase + 2, cbase + 1, 16);
        conv(inp, 3, 256, 256, pw, 16, ss, nullptr, 0, 0, convfull1, 16, 0, 1);
        pool(convfull1, 16, 256, 256, buf16_128);
        pw = packW(cbase + 6, 32, 16, 0); ss = foldBN(cbase + 8, cbase + 7, 32);
        conv(buf16_128, 16, 128, 128, pw, 32, ss, nullptr, 0, 0, convfull2, 32, 0, 1);
        pool(convfull2, 32, 128, 128, pooledB);
        pw = packW(cbase + 12, 16, 32, 0); ss = foldBN(cbase + 14, cbase + 13, 16);
        conv(pooledB, 32, 64, 64, pw, 16, ss, nullptr, 0, 0, c3out, 16, 0, 1);
        // resblock 1 (identity skip)
        pw = packW(bbase + 0, 16, 16, 0); ss = foldBN(bbase + 1, -1, 16);
        conv(c3out, 16, 64, 64, pw, 16, ss, nullptr, 0, 0, t0, 16, 0, 1);
        pw = packW(bbase + 5, 16, 16, 0); ss = foldBN(bbase + 6, -1, 16);
        conv(t0, 16, 64, 64, pw, 16, ss, c3out, 16, 0, t1, 16, 0, 1);
        // resblock 2 -> write into concat buffer at channel offset
        pw = packW(bbase + 10, 16, 16, 0); ss = foldBN(bbase + 11, -1, 16);
        conv(t1, 16, 64, 64, pw, 16, ss, nullptr, 0, 0, t0, 16, 0, 1);
        pw = packW(bbase + 15, 16, 16, 0); ss = foldBN(bbase + 16, -1, 16);
        conv(t0, 16, 64, 64, pw, 16, ss, t1, 16, 0, catb, 32, catOfs, 1);
    };

    branch(X, 2, 38, 0);
    branch(Y, 20, 58, 16);

    // b1c: 32 -> 16 with downsample skip conv
    PW pw; SS ss;
    pw = packW(78, 16, 32, 0); ss = foldBN(79, -1, 16);
    conv(catb, 32, 64, 64, pw, 16, ss, nullptr, 0, 0, t0, 16, 0, 1);
    pw = packW(88, 16, 32, 0); ss = foldBN(89, -1, 16);
    conv(catb, 32, 64, 64, pw, 16, ss, nullptr, 0, 0, t1, 16, 0, 0);  // skip path, no relu
    pw = packW(83, 16, 16, 0); ss = foldBN(84, -1, 16);
    conv(t0, 16, 64, 64, pw, 16, ss, t1, 16, 0, c3out, 16, 0, 1);
    // b2c
    pw = packW(93, 16, 16, 0); ss = foldBN(94, -1, 16);
    conv(c3out, 16, 64, 64, pw, 16, ss, nullptr, 0, 0, t0, 16, 0, 1);
    pw = packW(98, 16, 16, 0); ss = foldBN(99, -1, 16);
    conv(t0, 16, 64, 64, pw, 16, ss, c3out, 16, 0, t1, 16, 0, 1);

    // decoder: upsample -> convT(as flipped conv) -> BN -> ReLU, twice
    ups(t1, 16, 64, 64, buf16_128);
    pw = packW(103, 32, 16, 1); ss = foldBN(105, 104, 32);
    conv(buf16_128, 16, 128, 128, pw, 32, ss, nullptr, 0, 0, convfull2, 32, 0, 1);
    ups(convfull2, 32, 128, 128, bigU);
    pw = packW(109, 3, 32, 1); ss = foldBN(111, 110, 3);
    conv(bigU, 32, 256, 256, pw, 3, ss, nullptr, 0, 0, OUT, 3, 0, 1);
}
